// TagSelfAttentionEncoder_9861244911861
// MI455X (gfx1250) — matrix operand order repaired and checked by reading, not compiled, not run on hardware
//
#include <hip/hip_runtime.h>
#include <hip/hip_bf16.h>

// ---------------- types ----------------
typedef __attribute__((ext_vector_type(16))) __bf16 v16bf;
typedef __attribute__((ext_vector_type(8)))  float  v8f;

#define L_TOK   64
#define D_MODEL 512
#define N_HEAD  8
#define B_PER_WG 4
#define Y_ELEMS (4096u * 512u)   // y [B,512] precedes attn in d_out

__device__ __forceinline__ unsigned short f2bf(float f) {
  unsigned u = __float_as_uint(f);
  u += 0x7FFFu + ((u >> 16) & 1u);            // round-to-nearest-even
  return (unsigned short)(u >> 16);
}
__device__ __forceinline__ float bf2f(unsigned short h) {
  return __uint_as_float(((unsigned)h) << 16);
}

// WMMA A-operand (16-bit, 16x32) packed index (within one [32 lane][16 u16] tile set).
// Lane l: m=l%16, kh=l/16. VGPR v, half b: K = (v>=4?16:0) + (v&3)*2 + kh*8 + b
__device__ __forceinline__ int packA_idx(int grp, int mloc, int kt, int kk) {
  int hi = kk >> 4;
  int w  = kk & 15;
  int kh = w >> 3;
  int t  = w & 7;
  int v  = hi * 4 + (t >> 1);
  int hf = t & 1;
  int lane = kh * 16 + mloc;
  return ((grp * 16 + kt) * 32 + lane) * 16 + v * 2 + hf;
}

// -------- prep: pack w_q/w_k into WMMA-B bf16 layout; convert w_v/w_fc to plain bf16 --------
// d_ws layout (uint16 units):
//   [0       , 524288)  : Wq,Wk packed  ([sel][kt(16)][nt(32)][lane(32)][16])
//   [524288  , 786432)  : Wv  bf16 row-major [512][512]
//   [786432  , 1048576) : Wfc bf16 row-major [512][512]
extern "C" __global__ void __launch_bounds__(256)
pack_weights(const float* __restrict__ wq, const float* __restrict__ wk,
             const float* __restrict__ wv, const float* __restrict__ wfc,
             unsigned short* __restrict__ wsp) {
  int idx = blockIdx.x * 256 + threadIdx.x;   // 1048576 total
  if (idx < 524288) {
    int sel = idx >> 18;
    int r   = idx & 262143;
    int kt  = r >> 14;
    int r2  = r & 16383;
    int nt  = r2 >> 9;
    int r3  = r2 & 511;
    int ln  = r3 >> 4;
    int e   = r3 & 15;
    int v   = e >> 1, hf = e & 1;
    int n   = nt * 16 + (ln & 15);
    int K   = kt * 32 + (e >> 3) * 16 + (ln >> 4) * 8 + (e & 7);   // B-operand interleave
    const float* W = sel ? wk : wq;
    wsp[idx] = f2bf(W[(size_t)K * 512 + n]);
  } else {
    int j   = idx - 524288;
    int sel = j >> 18;                 // 0 = Wv, 1 = Wfc
    int pos = j & 262143;
    const float* W = sel ? wfc : wv;
    wsp[idx] = f2bf(W[pos]);
  }
}

// ---------------- fused kernel: one workgroup per 4 batch elements ----------------
// LDS map (bytes):
//   [0      ,262144) xA4   : packed bf16 x, 4 batch elems   (aliased by tail scratch)
//   [262144 ,278528) qc    : per-(h,dkt) Q slice, 4 b, A-layout   (aliased by xs)
//   [278528 ,294912) kc    : per-(h,dkt) K slice, 4 b, B-layout   (aliased by xs)
//   [294912 ,303104) asum  : [4][8][64] f32 attn column sums
extern "C" __global__ void __launch_bounds__(256)
fused_attn(const int* __restrict__ tags, const int* __restrict__ mask,
           const float* __restrict__ emb, const float* __restrict__ gamma,
           const float* __restrict__ beta,
           const unsigned short* __restrict__ wsp,
           float* __restrict__ out) {
  extern __shared__ char smem[];
  unsigned short* xA4  = (unsigned short*)(smem);            // 4 x 32768 u16
  unsigned short* qc   = (unsigned short*)(smem + 262144);   // [4][4 mg][32][16]
  unsigned short* kc   = (unsigned short*)(smem + 278528);   // [4][4 ng][32][16]
  float*          asum = (float*)(smem + 294912);            // [4][8][64]
  // tail aliases
  unsigned short* xs16   = (unsigned short*)(smem + 262144); // [4][8][512] bf16
  float*          pooled = (float*)(smem);                   // [4][512]
  float*          zbuf   = (float*)(smem + 8192);            // [4][512]
  float*          red    = (float*)(smem + 16384);           // [256]
  float*          red2   = (float*)(smem + 17408);           // [256]

  const unsigned short* wqk = wsp;
  const unsigned short* wvb = wsp + 524288;
  const unsigned short* wfb = wsp + 786432;

  const int tid  = threadIdx.x;
  const int lane = tid & 31;
  const int wid  = tid >> 5;
  const int b0   = blockIdx.x * B_PER_WG;

  // ---- zero asum ----
  for (int i = tid; i < 4 * 8 * 64; i += 256) asum[i] = 0.f;

  // ---- gather embeddings for all 4 b, convert bf16, pack WMMA-A layout ----
  for (int bs = 0; bs < B_PER_WG; ++bs) {
    const int b = b0 + bs;
    for (int idx = tid; idx < L_TOK * D_MODEL; idx += 256) {
      int row = idx >> 9, col = idx & 511;
      int tg  = tags[b * L_TOK + row];
      float v = emb[(size_t)tg * D_MODEL + col];
      xA4[bs * 32768 + packA_idx(row >> 4, row & 15, col >> 5, col & 31)] = f2bf(v);
    }
  }

  // ---- head loop: half-head-sliced projections + scores (weights read ONCE per WG) ----
  // GEMM waves : sel = wid>>2 (0=Q,1=K), mg = wid&3 (token group); all 4 b per weight load
  // Score waves: sb = wid>>1 (batch elem), hf2 = wid&1 (token-row half)
  const int sel = wid >> 2;
  const int mg  = wid & 3;
  const unsigned short* wpm = wqk + sel * 262144;
  unsigned short* dst = sel ? kc : qc;
  const float scale = sel ? 1.0f : 0.125f;   // fold 1/sqrt(DK) into Q

  const int sb  = wid >> 1;
  const int hf2 = wid & 1;
  const int kh  = lane >> 4;
  const int nlc = lane & 15;

  for (int h = 0; h < N_HEAD; ++h) {
    v8f SC[2][4];                      // score accumulators, persist across dkt
    #pragma unroll
    for (int ml = 0; ml < 2; ++ml)
      #pragma unroll
      for (int g = 0; g < 4; ++g) SC[ml][g] = (v8f){0.f,0.f,0.f,0.f,0.f,0.f,0.f,0.f};

    for (int dkt = 0; dkt < 2; ++dkt) {
      __syncthreads();                 // qc/kc free to rewrite

      // --- projection slice: 32 output cols (1 score k-tile) of Q and K, all 4 b ---
      {
        const int nt0 = h * 4 + dkt * 2;
        v8f C[4][2];
        #pragma unroll
        for (int bs = 0; bs < 4; ++bs)
          #pragma unroll
          for (int n = 0; n < 2; ++n) C[bs][n] = (v8f){0.f,0.f,0.f,0.f,0.f,0.f,0.f,0.f};
        for (int kt = 0; kt < 16; ++kt) {
          v16bf B0 = *(const v16bf*)(wpm + (((size_t)kt * 32 + nt0    ) * 32 + lane) * 16);
          v16bf B1 = *(const v16bf*)(wpm + (((size_t)kt * 32 + nt0 + 1) * 32 + lane) * 16);
          #pragma unroll
          for (int bs = 0; bs < 4; ++bs) {
            v16bf A = *(const v16bf*)(xA4 + bs * 32768 + ((mg * 16 + kt) * 32 + lane) * 16);
            C[bs][0] = __builtin_amdgcn_wmma_f32_16x16x32_bf16(
                           false, A, false, B0, (short)0, C[bs][0], false, false);
            C[bs][1] = __builtin_amdgcn_wmma_f32_16x16x32_bf16(
                           false, A, false, B1, (short)0, C[bs][1], false, false);
          }
        }
        // scatter slice into qc (A-layout) / kc (B-layout); kk = ntl*16 + nloc in [0,32)
        const int mbase = (lane >> 4) * 8;
        const int nloc  = lane & 15;
        #pragma unroll
        for (int ntl = 0; ntl < 2; ++ntl) {
          int laneQ = (nloc >> 3) * 16;                       // + mloc
          int eQ    = (ntl * 4 + ((nloc & 7) >> 1)) * 2 + (nloc & 1);
          int laneK = (nloc >> 3) * 16;                       // + mloc
          int eK    = ntl * 8 + (nloc & 7);
          int lsel  = sel ? laneK : laneQ;
          int esel  = sel ? eK    : eQ;
          #pragma unroll
          for (int bs = 0; bs < 4; ++bs) {
            #pragma unroll
            for (int r = 0; r < 8; ++r) {
              int mloc = mbase + r;
              dst[((bs * 4 + mg) * 32 + lsel + mloc) * 16 + esel] =
                  f2bf(C[bs][ntl][r] * scale);
            }
          }
        }
      }
      __syncthreads();                 // qc/kc ready

      // --- scores partial: accumulate q_slice . k_slice^T for this wave's rows ---
      #pragma unroll
      for (int ml = 0; ml < 2; ++ml) {
        int mgq = hf2 * 2 + ml;
        v16bf A = *(const v16bf*)(qc + ((sb * 4 + mgq) * 32 + lane) * 16);
        #pragma unroll
        for (int g = 0; g < 4; ++g) {
          v16bf B = *(const v16bf*)(kc + ((sb * 4 + g) * 32 + lane) * 16);
          SC[ml][g] = __builtin_amdgcn_wmma_f32_16x16x32_bf16(
                          false, A, false, B, (short)0, SC[ml][g], false, false);
        }
      }
    }

    // --- mask + softmax + attn writeback + asum for this head ---
    {
      const int b = b0 + sb;
      float ap[4] = {0.f, 0.f, 0.f, 0.f};
      #pragma unroll
      for (int ml = 0; ml < 2; ++ml) {
        int mgq = hf2 * 2 + ml;
        #pragma unroll
        for (int r = 0; r < 8; ++r) {
          int i = mgq * 16 + kh * 8 + r;
          #pragma unroll
          for (int g = 0; g < 4; ++g) {
            int j = g * 16 + nlc;
            int mv = __builtin_nontemporal_load(&mask[((size_t)b * 64 + i) * 64 + j]);
            if (mv == 0) SC[ml][g][r] = -1e9f;
          }
        }
        #pragma unroll
        for (int r = 0; r < 8; ++r) {
          float m0 = fmaxf(fmaxf(SC[ml][0][r], SC[ml][1][r]),
                           fmaxf(SC[ml][2][r], SC[ml][3][r]));
          #pragma unroll
          for (int s = 1; s < 16; s <<= 1) m0 = fmaxf(m0, __shfl_xor(m0, s, 32));
          float e[4], ssum = 0.f;
          #pragma unroll
          for (int g = 0; g < 4; ++g) { e[g] = __expf(SC[ml][g][r] - m0); ssum += e[g]; }
          #pragma unroll
          for (int s = 1; s < 16; s <<= 1) ssum += __shfl_xor(ssum, s, 32);
          float inv = 1.0f / ssum;
          int i = mgq * 16 + kh * 8 + r;
          size_t base = Y_ELEMS + (((size_t)b * N_HEAD + h) * 64 + i) * 64;
          #pragma unroll
          for (int g = 0; g < 4; ++g) {
            float a = e[g] * inv;
            ap[g] += a;
            __builtin_nontemporal_store(a, &out[base + g * 16 + nlc]);
          }
        }
      }
      #pragma unroll
      for (int g = 0; g < 4; ++g) {
        float t = ap[g] + __shfl_xor(ap[g], 16, 32);
        if (kh == 0)
          atomicAdd(&asum[(sb * 8 + h) * 64 + g * 16 + nlc], t);  // ds_add_f32
      }
    }
  }
  __syncthreads();

  // ---- xs[bs][h][d] = sum_j asum[bs][h][j] * x[bs][j][d]  (bf16, aliases qc/kc) ----
  for (int c = 0; c < 4; ++c) {
    int combo = tid + 256 * c;         // 1024 = 4 bs x 256 d-pairs
    int bs = combo >> 8;
    int d0 = (combo & 255) * 2, d1 = d0 + 1;
    const unsigned short* xb = xA4 + bs * 32768;
    float acc[N_HEAD][2];
    #pragma unroll
    for (int hh = 0; hh < N_HEAD; ++hh) { acc[hh][0] = 0.f; acc[hh][1] = 0.f; }
    for (int j = 0; j < L_TOK; ++j) {
      float x0 = bf2f(xb[packA_idx(j >> 4, j & 15, d0 >> 5, d0 & 31)]);
      float x1 = bf2f(xb[packA_idx(j >> 4, j & 15, d1 >> 5, d1 & 31)]);
      #pragma unroll
      for (int hh = 0; hh < N_HEAD; ++hh) {
        float a = asum[(bs * 8 + hh) * 64 + j];
        acc[hh][0] += a * x0;
        acc[hh][1] += a * x1;
      }
    }
    #pragma unroll
    for (int hh = 0; hh < N_HEAD; ++hh) {
      xs16[bs * 4096 + hh * 512 + d0] = f2bf(acc[hh][0]);
      xs16[bs * 4096 + hh * 512 + d1] = f2bf(acc[hh][1]);
    }
  }
  __syncthreads();

  // ---- pooled[bs][o] = xs[bs][o/64] . Wv[:, o]  (weights amortized over 4 b) ----
  for (int o = tid; o < D_MODEL; o += 256) {
    const int hh = o >> 6;
    float a0 = 0.f, a1 = 0.f, a2 = 0.f, a3 = 0.f;
    for (int d = 0; d < D_MODEL; ++d) {
      float w = bf2f(wvb[(size_t)d * 512 + o]);
      a0 += bf2f(xs16[0 * 4096 + hh * 512 + d]) * w;
      a1 += bf2f(xs16[1 * 4096 + hh * 512 + d]) * w;
      a2 += bf2f(xs16[2 * 4096 + hh * 512 + d]) * w;
      a3 += bf2f(xs16[3 * 4096 + hh * 512 + d]) * w;
    }
    pooled[0 * 512 + o] = a0; pooled[1 * 512 + o] = a1;
    pooled[2 * 512 + o] = a2; pooled[3 * 512 + o] = a3;
  }
  __syncthreads();

  // ---- z[bs][o] = pooled[bs] . w_fc[:, o] ----
  for (int o = tid; o < D_MODEL; o += 256) {
    float a0 = 0.f, a1 = 0.f, a2 = 0.f, a3 = 0.f;
    for (int d = 0; d < D_MODEL; ++d) {
      float w = bf2f(wfb[(size_t)d * 512 + o]);
      a0 += pooled[0 * 512 + d] * w; a1 += pooled[1 * 512 + d] * w;
      a2 += pooled[2 * 512 + d] * w; a3 += pooled[3 * 512 + d] * w;
    }
    zbuf[0 * 512 + o] = a0; zbuf[1 * 512 + o] = a1;
    zbuf[2 * 512 + o] = a2; zbuf[3 * 512 + o] = a3;
  }
  __syncthreads();

  // ---- layernorm over 512 per batch element, write y ----
  for (int bs = 0; bs < B_PER_WG; ++bs) {
    const float* zb = zbuf + bs * 512;
    float z0 = zb[tid], z1 = zb[tid + 256];
    red[tid]  = z0 + z1;
    red2[tid] = z0 * z0 + z1 * z1;
    __syncthreads();
    for (int s = 128; s > 0; s >>= 1) {
      if (tid < s) { red[tid] += red[tid + s]; red2[tid] += red2[tid + s]; }
      __syncthreads();
    }
    float mu   = red[0]  * (1.0f / 512.0f);
    float var  = red2[0] * (1.0f / 512.0f) - mu * mu;
    float rstd = __frsqrt_rn(var + 1e-6f);
    for (int o = tid; o < D_MODEL; o += 256)
      out[(size_t)(b0 + bs) * 512 + o] = (zb[o] - mu) * rstd * gamma[o] + beta[o];
    __syncthreads();
  }
}

// ---------------- launch ----------------
extern "C" void kernel_launch(void* const* d_in, const int* in_sizes, int n_in,
                              void* d_out, int out_size, void* d_ws, size_t ws_size,
                              hipStream_t stream) {
  const int*   tags  = (const int*)d_in[0];
  const int*   maskp = (const int*)d_in[1];
  const float* emb   = (const float*)d_in[2];
  const float* wq    = (const float*)d_in[3];
  const float* wk    = (const float*)d_in[4];
  const float* wv    = (const float*)d_in[5];
  const float* wfc   = (const float*)d_in[6];
  const float* gamma = (const float*)d_in[7];
  const float* beta  = (const float*)d_in[8];
  unsigned short* wsp = (unsigned short*)d_ws;   // 2 MB bf16 weight staging
  float* out = (float*)d_out;

  pack_weights<<<4096, 256, 0, stream>>>(wq, wk, wv, wfc, wsp);
  const size_t smem_bytes = 303104;              // ~296 KB < 320 KB WGP LDS
  fused_attn<<<1024, 256, smem_bytes, stream>>>(tags, maskp, emb,
                                                gamma, beta, wsp, out);
}
